// MultiHeadAttention_43026982371471
// MI455X (gfx1250) — compile-verified, matrix operand order checked by reading, not run on hardware
//
#include <hip/hip_runtime.h>

// Problem constants (reference: B=8, R=48, H=4, E=256 -> N=2304, D=64)
#define Bsz  8
#define Ntok 2304
#define Emb  256
#define Hh   4
#define Dd   64
#define Rr   48
#define JB   128   // staged KV tokens per block iteration (4 inner tiles of 32)

typedef __bf16 bf16_t;
typedef __attribute__((ext_vector_type(16))) __bf16 v16bf;
typedef __attribute__((ext_vector_type(8)))  __bf16 v8bf;
typedef __attribute__((ext_vector_type(8)))  float  v8f;

// Build a 16x32 (A) or 32x16 (B) bf16 WMMA fragment from two contiguous
// 16-byte runs: VGPRs 0-3 <- p0[0..7], VGPRs 4-7 <- p1[0..7].
static __device__ __forceinline__ v16bf load_frag(const bf16_t* p0, const bf16_t* p1) {
  v8bf lo = *(const v8bf*)p0;
  v8bf hi = *(const v8bf*)p1;
  return __builtin_shufflevector(lo, hi, 0,1,2,3,4,5,6,7,8,9,10,11,12,13,14,15);
}

static __device__ __forceinline__ v8f wmma_bf16(v16bf a, v16bf b, v8f c) {
  return __builtin_amdgcn_wmma_f32_16x16x32_bf16(false, a, false, b, (short)0, c, false, false);
}

static __device__ __forceinline__ int iabs(int v) { return v < 0 ? -v : v; }

// 16-byte async copy global -> LDS (ASYNCcnt-tracked, no VGPR data).
// Flat LDS pointers carry the LDS byte address in their low 32 bits.
static __device__ __forceinline__ void async_copy16(bf16_t* lds, const bf16_t* g) {
  unsigned loff = (unsigned)(uintptr_t)lds;
  unsigned long long ga = (unsigned long long)(uintptr_t)g;
  asm volatile("global_load_async_to_lds_b128 %0, %1, off"
               :: "v"(loff), "v"(ga) : "memory");
}

// Reductions across the 16 lanes sharing a C-tile row set (masks < 16 stay
// inside each half-wave group).
static __device__ __forceinline__ float redmax16(float v) {
  v = fmaxf(v, __shfl_xor(v, 1));
  v = fmaxf(v, __shfl_xor(v, 2));
  v = fmaxf(v, __shfl_xor(v, 4));
  v = fmaxf(v, __shfl_xor(v, 8));
  return v;
}
static __device__ __forceinline__ float redsum16(float v) {
  v += __shfl_xor(v, 1);
  v += __shfl_xor(v, 2);
  v += __shfl_xor(v, 4);
  v += __shfl_xor(v, 8);
  return v;
}

// ---------------- Kernel 0: f32 -> bf16 conversion ----------------
__global__ void cvt_f32_bf16(const float* __restrict__ src, bf16_t* __restrict__ dst, int n) {
  for (int i = blockIdx.x * blockDim.x + threadIdx.x; i < n; i += gridDim.x * blockDim.x)
    dst[i] = (bf16_t)src[i];
}

// ---------------- Kernel 1: QKV projection (bf16 WMMA GEMM) ----------------
// xb [B*N, 256] row-major bf16; Wb = Wqkv [768, 256] row-major bf16.
// Each wave computes a 16x64 tile of x @ Wqkv^T, then scatters into:
//   Qs [b,h,tok,d] (pre-scaled by D^-0.5), Kb [b,h,tok,d], Vt [b,h,d,tok].
__global__ __launch_bounds__(256) void qkv_gemm(
    const bf16_t* __restrict__ xb, const bf16_t* __restrict__ Wb,
    bf16_t* __restrict__ Qs, bf16_t* __restrict__ Kb, bf16_t* __restrict__ Vt)
{
  const int wid  = blockIdx.x * 8 + (threadIdx.x >> 5);
  const int lane = threadIdx.x & 31;
  const int lh = lane >> 4, ln = lane & 15;
  const int mtile = wid / 12;      // 1152 row tiles of 16
  const int ntile = wid % 12;      // 12 col tiles of 64 (768 total)
  const int m = mtile * 16 + ln;

  v8f acc[4] = {};
  for (int c = 0; c < 8; ++c) {    // K = 256, chunks of 32
    const bf16_t* arow = xb + (size_t)m * Emb + c * 32 + lh * 8;
    v16bf a = load_frag(arow, arow + 16);
#pragma unroll
    for (int s = 0; s < 4; ++s) {
      const int co = ntile * 64 + s * 16 + ln;   // output column = W row
      const bf16_t* brow = Wb + (size_t)co * Emb + c * 32 + lh * 8;
      acc[s] = wmma_bf16(a, load_frag(brow, brow + 16), acc[s]);
    }
  }

#pragma unroll
  for (int s = 0; s < 4; ++s) {
    const int co  = ntile * 64 + s * 16 + ln;
    const int sel = co >> 8;        // 0=Q 1=K 2=V
    const int cc  = co & 255;
    const int hd  = cc >> 6;
    const int d   = cc & 63;
#pragma unroll
    for (int r = 0; r < 8; ++r) {
      const int row = mtile * 16 + lh * 8 + r;   // flat b*N + tok
      const int b   = row / Ntok;
      const int tok = row % Ntok;
      const float v = acc[s][r];
      const size_t qkIdx = (((size_t)(b * Hh + hd)) * Ntok + tok) * Dd + d;
      if (sel == 0)      Qs[qkIdx] = (bf16_t)(v * 0.125f);        // scale = D^-0.5
      else if (sel == 1) Kb[qkIdx] = (bf16_t)v;
      else Vt[(((size_t)(b * Hh + hd)) * Dd + d) * Ntok + tok] = (bf16_t)v;
    }
  }
}

// ---------------- Kernel 2: flash attention with async-LDS KV staging -------
// Grid: B*H*18 blocks of 256 threads (8 waves). Each wave owns a 16-row Q
// strip. K/V are staged cooperatively into double-buffered LDS super-tiles of
// 128 tokens via global_load_async_to_lds_b128 (one barrier per super-tile),
// cutting L2 read traffic 8x vs per-wave streaming. Rel-pos bias is computed
// analytically and gathered from an LDS-resident per-head bias row.
__global__ __launch_bounds__(256) void flash_attn(
    const bf16_t* __restrict__ Qs, const bf16_t* __restrict__ Kb,
    const bf16_t* __restrict__ Vt, const float* __restrict__ biases,
    bf16_t* __restrict__ AO)
{
  __shared__ float biasRow[Ntok];                         // 9216 B
  __shared__ __align__(16) bf16_t Ktile[2][JB][Dd];       // 2 x 16 KB
  __shared__ __align__(16) bf16_t Vtile[2][Dd][JB];       // 2 x 16 KB
  __shared__ __align__(16) bf16_t Pbuf[8][16 * 32];       // 8 KB

  const int bh  = blockIdx.x / 18;
  const int blk = blockIdx.x % 18;
  const int b = bh >> 2;
  const int h = bh & 3;

  const int wave = threadIdx.x >> 5;
  const int lane = threadIdx.x & 31;
  const int lh = lane >> 4, ln = lane & 15;
  const int i0 = blk * 128 + wave * 16;

  const bf16_t* Qbase = Qs + (size_t)bh * Ntok * Dd;
  const bf16_t* Kbase = Kb + (size_t)bh * Ntok * Dd;
  const bf16_t* Vbase = Vt + (size_t)bh * Dd * Ntok;

  for (int i = threadIdx.x; i < Ntok; i += 256) biasRow[i] = biases[h * Ntok + i];

  // Stage KV super-tile jb into LDS parity buffer (8 async b128 per lane).
  auto stage_kv = [&](int jb, int p) {
    // K: 128 rows x 64 d, contiguous 16 KB; this wave copies 16 rows (2 KB)
    {
      const bf16_t* src = Kbase + ((size_t)jb * JB + wave * 16) * Dd;
      bf16_t* dst = &Ktile[p][wave * 16][0];
#pragma unroll
      for (int u = 0; u < 4; ++u) {
        const int off = (u * 32 + lane) * 8;          // 16B chunks, lane-contig
        async_copy16(dst + off, src + off);
      }
    }
    // V: 64 d-rows x 128 tokens from Vt (global row stride Ntok); 8 rows/wave
    {
#pragma unroll
      for (int u = 0; u < 4; ++u) {
        const int k   = u * 32 + lane;                // 0..127 chunk id
        const int row = k >> 4, c16 = k & 15;
        const bf16_t* src = Vbase + (size_t)(wave * 8 + row) * Ntok
                                  + (size_t)jb * JB + c16 * 8;
        bf16_t* dst = &Vtile[p][wave * 8 + row][c16 * 8];
        async_copy16(dst, src);
      }
    }
  };

  // Q fragments for both 32-wide d-chunks (held for whole loop)
  v16bf qf0, qf1;
  {
    const bf16_t* qrow = Qbase + (size_t)(i0 + ln) * Dd + lh * 8;
    qf0 = load_frag(qrow, qrow + 16);
    qf1 = load_frag(qrow + 32, qrow + 48);
  }

  float mrun[8], lrun[8];
  int ri[8], ci[8];
#pragma unroll
  for (int r = 0; r < 8; ++r) {
    mrun[r] = -1e30f; lrun[r] = 0.f;
    const int irow = i0 + lh * 8 + r;
    ri[r] = irow / Rr; ci[r] = irow % Rr;
  }
  v8f o0 = {}, o1 = {}, o2 = {}, o3 = {};

  // Prologue: stage block 0; wait own async ops, then block barrier (also
  // covers the biasRow fill).
  stage_kv(0, 0);
  asm volatile("s_wait_asynccnt 0x0" ::: "memory");
  __syncthreads();

  for (int jb = 0; jb < Ntok / JB; ++jb) {
    const int p = jb & 1;
    if (jb + 1 < Ntok / JB) stage_kv(jb + 1, p ^ 1);   // prefetch next buffer

    const bf16_t* Kl = &Ktile[p][0][0];
    const bf16_t* Vl = &Vtile[p][0][0];

    for (int jt = 0; jt < 4; ++jt) {
      const int jl  = jt * 32;            // local col in staged tiles
      const int j0g = jb * JB + jl;       // global col of subtile 0

      // S = (Q*scale) @ K^T for two 16-col subtiles (K frags from LDS)
      v8f s0 = {}, s1 = {};
      {
        const bf16_t* krow = Kl + (size_t)(jl + ln) * Dd + lh * 8;
        v16bf k0 = load_frag(krow, krow + 16);
        v16bf k1 = load_frag(krow + 32, krow + 48);
        s0 = wmma_bf16(qf0, k0, s0);
        s0 = wmma_bf16(qf1, k1, s0);
      }
      {
        const bf16_t* krow = Kl + (size_t)(jl + 16 + ln) * Dd + lh * 8;
        v16bf k0 = load_frag(krow, krow + 16);
        v16bf k1 = load_frag(krow + 32, krow + 48);
        s1 = wmma_bf16(qf0, k0, s1);
        s1 = wmma_bf16(qf1, k1, s1);
      }

      // analytic rel-pos bias: idx = |ri-rj|*48 + |ci-cj|
      const int j0c = j0g + ln;        const int rj0 = j0c / Rr, cj0 = j0c % Rr;
      const int j1c = j0g + 16 + ln;   const int rj1 = j1c / Rr, cj1 = j1c % Rr;
      float sm[8];
#pragma unroll
      for (int r = 0; r < 8; ++r) {
        const int d0 = iabs(ri[r] - rj0) * Rr + iabs(ci[r] - cj0);
        const int d1 = iabs(ri[r] - rj1) * Rr + iabs(ci[r] - cj1);
        const float a0 = s0[r] + biasRow[d0];
        const float a1 = s1[r] + biasRow[d1];
        s0[r] = a0; s1[r] = a1;
        sm[r] = fmaxf(a0, a1);
      }

      // online softmax update
      float mnew[8], corr[8], psum[8];
#pragma unroll
      for (int r = 0; r < 8; ++r) {
        const float rm = redmax16(sm[r]);
        mnew[r] = fmaxf(mrun[r], rm);
        corr[r] = __expf(mrun[r] - mnew[r]);
      }
#pragma unroll
      for (int r = 0; r < 8; ++r) {
        const float p0 = __expf(s0[r] - mnew[r]);
        const float p1 = __expf(s1[r] - mnew[r]);
        psum[r] = p0 + p1;
        Pbuf[wave][(lh * 8 + r) * 32 + ln]      = (bf16_t)p0;  // C-layout -> LDS
        Pbuf[wave][(lh * 8 + r) * 32 + 16 + ln] = (bf16_t)p1;
      }
#pragma unroll
      for (int r = 0; r < 8; ++r) {
        const float rs = redsum16(psum[r]);
        lrun[r] = lrun[r] * corr[r] + rs;
        mrun[r] = mnew[r];
        o0[r] *= corr[r]; o1[r] *= corr[r]; o2[r] *= corr[r]; o3[r] *= corr[r];
      }

      // same-wave LDS transpose round-trip: DS ops are in-order per wave;
      // wait for the b16 stores, then read back in A-matrix layout.
      asm volatile("s_wait_dscnt 0x0" ::: "memory");
      v16bf pf;
      {
        const bf16_t* prow = &Pbuf[wave][ln * 32 + lh * 8];
        pf = load_frag(prow, prow + 16);
      }
      asm volatile("" ::: "memory");  // keep later stores after these loads

      // O += P @ V  (V frags from LDS, token runs contiguous)
      {
        const bf16_t* vr0 = Vl + (size_t)(0 * 16 + ln) * JB + jl + lh * 8;
        o0 = wmma_bf16(pf, load_frag(vr0, vr0 + 16), o0);
        const bf16_t* vr1 = Vl + (size_t)(1 * 16 + ln) * JB + jl + lh * 8;
        o1 = wmma_bf16(pf, load_frag(vr1, vr1 + 16), o1);
        const bf16_t* vr2 = Vl + (size_t)(2 * 16 + ln) * JB + jl + lh * 8;
        o2 = wmma_bf16(pf, load_frag(vr2, vr2 + 16), o2);
        const bf16_t* vr3 = Vl + (size_t)(3 * 16 + ln) * JB + jl + lh * 8;
        o3 = wmma_bf16(pf, load_frag(vr3, vr3 + 16), o3);
      }
    }

    // own async prefetch done -> barrier makes next buffer globally ready,
    // and guarantees no wave still reads the buffer we overwrite next.
    asm volatile("s_wait_asynccnt 0x0" ::: "memory");
    __syncthreads();
  }

  // normalize and write attn output bf16 in [b, tok, h*64+d] layout
#pragma unroll
  for (int r = 0; r < 8; ++r) {
    const int tok = i0 + lh * 8 + r;
    const float inv = 1.f / lrun[r];
    const size_t base = ((size_t)(b * Ntok + tok)) * Emb + h * Dd;
    AO[base + 0 * 16 + ln] = (bf16_t)(o0[r] * inv);
    AO[base + 1 * 16 + ln] = (bf16_t)(o1[r] * inv);
    AO[base + 2 * 16 + ln] = (bf16_t)(o2[r] * inv);
    AO[base + 3 * 16 + ln] = (bf16_t)(o3[r] * inv);
  }
}

// ---------------- Kernel 3: output projection + bias ----------------
__global__ __launch_bounds__(256) void out_gemm(
    const bf16_t* __restrict__ AO, const bf16_t* __restrict__ Wob,
    const float* __restrict__ bout, float* __restrict__ out)
{
  const int wid  = blockIdx.x * 8 + (threadIdx.x >> 5);
  const int lane = threadIdx.x & 31;
  const int lh = lane >> 4, ln = lane & 15;
  const int mtile = wid >> 2;      // 1152 row tiles
  const int ntile = wid & 3;       // 4 col tiles of 64 (256 total)
  const int m = mtile * 16 + ln;

  v8f acc[4] = {};
  for (int c = 0; c < 8; ++c) {
    const bf16_t* arow = AO + (size_t)m * Emb + c * 32 + lh * 8;
    v16bf a = load_frag(arow, arow + 16);
#pragma unroll
    for (int s = 0; s < 4; ++s) {
      const int co = ntile * 64 + s * 16 + ln;
      const bf16_t* brow = Wob + (size_t)co * Emb + c * 32 + lh * 8;
      acc[s] = wmma_bf16(a, load_frag(brow, brow + 16), acc[s]);
    }
  }
#pragma unroll
  for (int s = 0; s < 4; ++s) {
    const int co = ntile * 64 + s * 16 + ln;
    const float bb = bout[co];
#pragma unroll
    for (int r = 0; r < 8; ++r) {
      const int row = mtile * 16 + lh * 8 + r;
      out[(size_t)row * Emb + co] = acc[s][r] + bb;
    }
  }
}

// ---------------- Launch ----------------
extern "C" void kernel_launch(void* const* d_in, const int* in_sizes, int n_in,
                              void* d_out, int out_size, void* d_ws, size_t ws_size,
                              hipStream_t stream) {
  const float* x      = (const float*)d_in[0];   // [8,2304,256]
  const float* Wqkv   = (const float*)d_in[1];   // [768,256]
  const float* Wout   = (const float*)d_in[2];   // [256,256]
  const float* bout   = (const float*)d_in[3];   // [256]
  const float* biases = (const float*)d_in[4];   // [4,2304]
  // d_in[5] = bias_idxs: unused — index is computed analytically in-kernel.
  (void)in_sizes; (void)n_in; (void)out_size; (void)ws_size;

  char* ws = (char*)d_ws;
  const size_t SZ_XM  = (size_t)Bsz * Ntok * Emb * sizeof(bf16_t);      // 9.44 MB
  const size_t SZ_WQ  = (size_t)3 * Emb * Emb * sizeof(bf16_t);         // 393 KB
  const size_t SZ_WO  = (size_t)Emb * Emb * sizeof(bf16_t);             // 131 KB
  const size_t SZ_QKV = (size_t)Bsz * Hh * Ntok * Dd * sizeof(bf16_t);  // 9.44 MB

  size_t off = 0;
  bf16_t* xb   = (bf16_t*)(ws + off); off += SZ_XM;
  bf16_t* Wqb  = (bf16_t*)(ws + off); off += SZ_WQ;
  bf16_t* Wob  = (bf16_t*)(ws + off); off += SZ_WO;
  bf16_t* Qs   = (bf16_t*)(ws + off); off += SZ_QKV;
  bf16_t* Kb   = (bf16_t*)(ws + off); off += SZ_QKV;
  bf16_t* Vt   = (bf16_t*)(ws + off); off += SZ_QKV;
  bf16_t* AO   = (bf16_t*)(ws + off); off += SZ_XM;   // ~47.7 MB total

  cvt_f32_bf16<<<2048, 256, 0, stream>>>(x,    xb,  Bsz * Ntok * Emb);
  cvt_f32_bf16<<<192,  256, 0, stream>>>(Wqkv, Wqb, 3 * Emb * Emb);
  cvt_f32_bf16<<<64,   256, 0, stream>>>(Wout, Wob, Emb * Emb);

  // 1152 M-tiles * 12 N-tiles = 13824 waves / 8 per block
  qkv_gemm<<<1728, 256, 0, stream>>>(xb, Wqb, Qs, Kb, Vt);

  // B*H*18 row-blocks
  flash_attn<<<Bsz * Hh * 18, 256, 0, stream>>>(Qs, Kb, Vt, biases, AO);

  // 1152 M-tiles * 4 N-tiles = 4608 waves / 8 per block
  out_gemm<<<576, 256, 0, stream>>>(AO, Wob, bout, (float*)d_out);
}
